// TransformerEncoderLayer_26001732010564
// MI455X (gfx1250) — matrix operand order repaired and checked by reading, not compiled, not run on hardware
//
#include <hip/hip_runtime.h>
#include <hip/hip_bf16.h>
#include <math.h>

typedef __bf16 bf16;
typedef __attribute__((ext_vector_type(8)))  bf16  bf16x8;
typedef __attribute__((ext_vector_type(16))) bf16  bf16x16;
typedef __attribute__((ext_vector_type(8)))  float f32x8;

#define D_MODEL 1024
#define D_FF    4096
#define NHEAD   16
#define DKH     64
#define SEQ     2048
#define BATCH   2

// ---------------------------------------------------------------------------
// CDNA5 async global->LDS copy (ASYNCcnt path) via inline asm, with fallback
// ---------------------------------------------------------------------------
#if defined(__gfx1250__)
#define HAVE_ASYNC_LDS 1
#else
#define HAVE_ASYNC_LDS 0
#endif

static __device__ __forceinline__ void stage16(void* lds, const void* g) {
#if HAVE_ASYNC_LDS
  // per-lane 16B copy: LDS[vdst] = MEM[vaddr]; tracked by ASYNCcnt
  unsigned loff = (unsigned)(size_t)(__attribute__((address_space(3))) void*)lds;
  unsigned long long gaddr = (unsigned long long)(size_t)g;
  asm volatile("global_load_async_to_lds_b128 %0, %1, off"
               :: "v"(loff), "v"(gaddr)
               : "memory");
#else
  *(bf16x8*)lds = *(const bf16x8*)g;
#endif
}
static __device__ __forceinline__ void wait_async() {
#if HAVE_ASYNC_LDS
#if __has_builtin(__builtin_amdgcn_s_wait_asynccnt)
  __builtin_amdgcn_s_wait_asynccnt(0);
#else
  asm volatile("s_wait_asynccnt 0x0" ::: "memory");
#endif
#endif
}

static __device__ __forceinline__ bf16x16 cat8(bf16x8 lo, bf16x8 hi) {
  return __builtin_shufflevector(lo, hi, 0,1,2,3,4,5,6,7,8,9,10,11,12,13,14,15);
}
static __device__ __forceinline__ f32x8 zero8() {
  f32x8 z;
#pragma unroll
  for (int i = 0; i < 8; ++i) z[i] = 0.f;
  return z;
}

// ---------------------------------------------------------------------------
// fp32 -> bf16 conversion
// ---------------------------------------------------------------------------
__global__ void f32_to_bf16(const float* __restrict__ in, bf16* __restrict__ out, long n) {
  long i = (long)blockIdx.x * blockDim.x + threadIdx.x;
  long stride = (long)gridDim.x * blockDim.x;
  for (; i < n; i += stride) out[i] = (bf16)in[i];
}

// ---------------------------------------------------------------------------
// bf16 GEMM, f32 accumulate: C[M,N] = act(A[M,K] @ B[K,N] + bias[N])
// ACT: 0 = identity, 1 = exact GELU, 2 = scale by 1/sqrt(dk) (for Q proj)
// 256 threads = 8 waves; block tile 128x128; wave tile 64x32 (4x2 WMMA frags)
// Double-buffered LDS; A staged via async global->LDS; B staged transposed
// (N-major) so B-frags are contiguous ds_load_b128 pairs.
// ---------------------------------------------------------------------------
#define BM 128
#define BN 128
#define BK 32
#define BSTR (BK + 16)   // 48 elem = 96B row stride (16B aligned)

template<int ACT, bool HASF, bool HASB>
__global__ __launch_bounds__(256)
void gemm_bf16_wmma(const bf16* __restrict__ A, const bf16* __restrict__ Bw,
                    const float* __restrict__ bias,
                    float* __restrict__ outF, bf16* __restrict__ outB,
                    int M, int N, int K)
{
  __shared__ bf16 As [2][BM][BSTR];   // row-major A tiles
  __shared__ bf16 BsT[2][BN][BSTR];   // transposed B tiles: [n][k]

  const int tid  = threadIdx.x;
  const int lane = tid & 31;
  const int wave = tid >> 5;   // 0..7
  const int wm   = wave >> 2;  // 0..1  (64-row slab)
  const int wn   = wave & 3;   // 0..3  (32-col slab)
  const int bm   = blockIdx.y * BM;
  const int bn   = blockIdx.x * BN;

  f32x8 acc[4][2];
#pragma unroll
  for (int i = 0; i < 4; ++i)
#pragma unroll
    for (int j = 0; j < 2; ++j) acc[i][j] = zero8();

  const int arow = tid >> 1, acol = (tid & 1) * 16;  // A tile: 128 x 32
  const int brow = tid >> 3, bcol = (tid & 7) * 16;  // B tile: 32 x 128
  const int l15  = lane & 15;
  const int kh   = (lane >> 4) * 8;    // A-frag K-half select
  const int kb   = (lane >> 4) * 8;    // B-frag K-half select

  auto stageA = [&](int buf, int k0) {
    const bf16* gA = A + (size_t)(bm + arow) * K + k0 + acol;
    stage16(&As[buf][arow][acol],     gA);
    stage16(&As[buf][arow][acol + 8], gA + 8);
  };
  auto stageB = [&](int buf, int k0) {
    const bf16* gB = Bw + (size_t)(k0 + brow) * N + bn + bcol;
    bf16x8 b0 = *(const bf16x8*)(gB);
    bf16x8 b1 = *(const bf16x8*)(gB + 8);
    if (k0 + 2 * BK < K) __builtin_prefetch(gB + (size_t)BK * N, 0, 0);
#pragma unroll
    for (int e = 0; e < 8; ++e) BsT[buf][bcol + e][brow]     = b0[e];
#pragma unroll
    for (int e = 0; e < 8; ++e) BsT[buf][bcol + 8 + e][brow] = b1[e];
  };

  stageA(0, 0);
  stageB(0, 0);
  wait_async();
  __syncthreads();

  int cur = 0;
  for (int k0 = 0; k0 < K; k0 += BK) {
    const int nxt = cur ^ 1;
    if (k0 + BK < K) { stageA(nxt, k0 + BK); stageB(nxt, k0 + BK); }

    // A fragments: two contiguous b128 per frag
    bf16x16 af[4];
#pragma unroll
    for (int i = 0; i < 4; ++i) {
      const bf16* p = &As[cur][wm * 64 + i * 16 + l15][0];
      af[i] = cat8(*(const bf16x8*)(p + kh), *(const bf16x8*)(p + 16 + kh));
    }
    // B fragments from transposed tile: two contiguous b128 per frag
    bf16x16 bfr[2];
#pragma unroll
    for (int j = 0; j < 2; ++j) {
      const bf16* p = &BsT[cur][wn * 32 + j * 16 + l15][0];
      bfr[j] = cat8(*(const bf16x8*)(p + kb), *(const bf16x8*)(p + kb + 16));
    }
#pragma unroll
    for (int i = 0; i < 4; ++i)
#pragma unroll
      for (int j = 0; j < 2; ++j)
        acc[i][j] = __builtin_amdgcn_wmma_f32_16x16x32_bf16(
            false, af[i], false, bfr[j], (short)0, acc[i][j], false, false);

    wait_async();
    __syncthreads();
    cur = nxt;
  }

  // epilogue: C layout — lane = col, VGPR r = row (lanes>=16 -> +8)
  const int rbase = (lane >> 4) * 8;
#pragma unroll
  for (int i = 0; i < 4; ++i) {
#pragma unroll
    for (int j = 0; j < 2; ++j) {
      const int col = bn + wn * 32 + j * 16 + l15;
      const float bv = bias[col];
#pragma unroll
      for (int r = 0; r < 8; ++r) {
        const int row = bm + wm * 64 + i * 16 + rbase + r;
        float v = acc[i][j][r] + bv;
        if (ACT == 1) v = 0.5f * v * (1.0f + erff(v * 0.70710678118654752f));
        if (ACT == 2) v *= 0.125f;   // fold 1/sqrt(dk) into Q projection
        const size_t idx = (size_t)row * N + col;
        if (HASF) outF[idx] = v;
        if (HASB) outB[idx] = (bf16)v;
      }
    }
  }
}

// ---------------------------------------------------------------------------
// Flash attention: block = 4 waves, 64 query rows, one (b,h) per blockIdx.y.
// Q is pre-scaled by 1/sqrt(dk) in the projection GEMM, so the score tiles
// come out of WMMA ready for softmax. K tile staged async (row-major,
// contiguous frag reads); V tile staged transposed (dk-major).
// ---------------------------------------------------------------------------
__global__ __launch_bounds__(128)
void attention_wmma(const bf16* __restrict__ Q, const bf16* __restrict__ Km,
                    const bf16* __restrict__ V, bf16* __restrict__ ctx,
                    int S, int D, int H)
{
  __shared__ bf16 Kt [64][72];      // [key][dk]
  __shared__ bf16 VtT[64][72];      // [dk][key]  (transposed)
  __shared__ bf16 Pst[4][16][72];   // per-wave P staging (C layout -> A layout)

  const int tid  = threadIdx.x;
  const int lane = tid & 31;
  const int wave = tid >> 5;          // 0..3
  const int bh   = blockIdx.y;
  const int b    = bh / H;
  const int h    = bh % H;
  const int q0   = blockIdx.x * 64 + wave * 16;
  const int l15  = lane & 15;
  const int kh   = (lane >> 4) * 8;
  const int kb   = (lane >> 4) * 8;

  // Q fragments (A layout) straight from global, dk=64 -> 2 K-chunks
  const bf16* qrow = Q + ((size_t)b * S + q0 + l15) * D + h * DKH;
  bf16x16 qf[2];
#pragma unroll
  for (int c = 0; c < 2; ++c) {
    bf16x8 lo = *(const bf16x8*)(qrow + c * 32 + kh);
    bf16x8 hi = *(const bf16x8*)(qrow + c * 32 + 16 + kh);
    qf[c] = cat8(lo, hi);
  }

  f32x8 o[4];
#pragma unroll
  for (int j = 0; j < 4; ++j) o[j] = zero8();
  float mrow[8], lrow[8];
#pragma unroll
  for (int r = 0; r < 8; ++r) { mrow[r] = -1e30f; lrow[r] = 0.f; }

  for (int kt = 0; kt < S / 64; ++kt) {
    {  // cooperative staging: K async (no transpose), V transposed via VGPRs
      const int r  = tid >> 1;
      const int c0 = (tid & 1) * 32;
      const bf16* gK = Km + ((size_t)b * S + kt * 64 + r) * D + h * DKH + c0;
      const bf16* gV = V  + ((size_t)b * S + kt * 64 + r) * D + h * DKH + c0;
#pragma unroll
      for (int u = 0; u < 4; ++u) stage16(&Kt[r][c0 + u * 8], gK + u * 8);
#pragma unroll
      for (int u = 0; u < 4; ++u) {
        bf16x8 vv = *(const bf16x8*)(gV + u * 8);
#pragma unroll
        for (int e = 0; e < 8; ++e) VtT[c0 + u * 8 + e][r] = vv[e];
      }
    }
    wait_async();
    __syncthreads();

    // S = Q @ K^T : 16 x 64; B-frag = contiguous dk run of one key row
    f32x8 sc[4];
#pragma unroll
    for (int j = 0; j < 4; ++j) {
      f32x8 s = zero8();
#pragma unroll
      for (int c = 0; c < 2; ++c) {
        const bf16* kp = &Kt[j * 16 + l15][c * 32 + kb];
        bf16x16 bf = cat8(*(const bf16x8*)(kp), *(const bf16x8*)(kp + 16));
        s = __builtin_amdgcn_wmma_f32_16x16x32_bf16(
            false, qf[c], false, bf, (short)0, s, false, false);
      }
      sc[j] = s;
    }

    // online softmax row stats (rows live per-VGPR; 16-lane reductions)
    float mnew[8], alpha[8];
#pragma unroll
    for (int r = 0; r < 8; ++r) {
      float v = sc[0][r];
#pragma unroll
      for (int j = 1; j < 4; ++j) v = fmaxf(v, sc[j][r]);
#pragma unroll
      for (int msk = 1; msk < 16; msk <<= 1) v = fmaxf(v, __shfl_xor(v, msk, 32));
      mnew[r]  = fmaxf(mrow[r], v);
      alpha[r] = __expf(mrow[r] - mnew[r]);
      mrow[r]  = mnew[r];
    }
#pragma unroll
    for (int j = 0; j < 4; ++j)
#pragma unroll
      for (int r = 0; r < 8; ++r) sc[j][r] = __expf(sc[j][r] - mnew[r]);
#pragma unroll
    for (int r = 0; r < 8; ++r) {
      float s = 0.f;
#pragma unroll
      for (int j = 0; j < 4; ++j) s += sc[j][r];
#pragma unroll
      for (int msk = 1; msk < 16; msk <<= 1) s += __shfl_xor(s, msk, 32);
      lrow[r] = lrow[r] * alpha[r] + s;
    }

    // stage P (C layout) -> LDS, reload as A layout for P@V
    const int prow = (lane >> 4) * 8;
#pragma unroll
    for (int j = 0; j < 4; ++j)
#pragma unroll
      for (int r = 0; r < 8; ++r)
        Pst[wave][prow + r][j * 16 + l15] = (bf16)sc[j][r];

#pragma unroll
    for (int j = 0; j < 4; ++j)
#pragma unroll
      for (int r = 0; r < 8; ++r) o[j][r] *= alpha[r];

    bf16x16 pa[2];
#pragma unroll
    for (int c = 0; c < 2; ++c) {
      const bf16* pp = &Pst[wave][l15][c * 32 + kh];
      pa[c] = cat8(*(const bf16x8*)(pp), *(const bf16x8*)(pp + 16));
    }
#pragma unroll
    for (int j = 0; j < 4; ++j) {
#pragma unroll
      for (int c = 0; c < 2; ++c) {
        const bf16* vp = &VtT[j * 16 + l15][c * 32 + kb];
        bf16x16 vbv = cat8(*(const bf16x8*)(vp), *(const bf16x8*)(vp + 16));
        o[j] = __builtin_amdgcn_wmma_f32_16x16x32_bf16(
            false, pa[c], false, vbv, (short)0, o[j], false, false);
      }
    }
    __syncthreads();
  }

  // finalize and store context (bf16, ready for Wo GEMM)
  const int rbase = (lane >> 4) * 8;
#pragma unroll
  for (int r = 0; r < 8; ++r) {
    const float inv = 1.0f / lrow[r];
    const int row = q0 + rbase + r;
    bf16* crow = ctx + ((size_t)b * S + row) * D + h * DKH;
#pragma unroll
    for (int j = 0; j < 4; ++j) crow[j * 16 + l15] = (bf16)(o[j][r] * inv);
  }
}

// ---------------------------------------------------------------------------
// fused residual-add + LayerNorm; optional bf16 shadow copy for next GEMM
// ---------------------------------------------------------------------------
__global__ __launch_bounds__(256)
void add_layernorm(const float* __restrict__ A, const float* __restrict__ Bv,
                   const float* __restrict__ g, const float* __restrict__ be,
                   float* __restrict__ outF, bf16* __restrict__ outB, int D)
{
  __shared__ float red[512];
  const int row = blockIdx.x;
  const int t = threadIdx.x;
  float v[4];
  float s = 0.f, sq = 0.f;
#pragma unroll
  for (int e = 0; e < 4; ++e) {
    const int c = t + e * 256;
    const size_t idx = (size_t)row * D + c;
    v[e] = A[idx] + Bv[idx];
    s += v[e]; sq += v[e] * v[e];
  }
  red[t] = s; red[256 + t] = sq;
  __syncthreads();
  for (int off = 128; off > 0; off >>= 1) {
    if (t < off) { red[t] += red[t + off]; red[256 + t] += red[256 + t + off]; }
    __syncthreads();
  }
  const float mu   = red[0] / D;
  const float var  = red[256] / D - mu * mu;
  const float rstd = rsqrtf(var + 1e-5f);
#pragma unroll
  for (int e = 0; e < 4; ++e) {
    const int c = t + e * 256;
    const size_t idx = (size_t)row * D + c;
    const float y = (v[e] - mu) * rstd * g[c] + be[c];
    if (outF) outF[idx] = y;
    if (outB) outB[idx] = (bf16)y;
  }
}

// ---------------------------------------------------------------------------
// host-side orchestration
// ---------------------------------------------------------------------------
extern "C" void kernel_launch(void* const* d_in, const int* in_sizes, int n_in,
                              void* d_out, int out_size, void* d_ws, size_t ws_size,
                              hipStream_t stream)
{
  const float* x   = (const float*)d_in[0];
  const float* Wq  = (const float*)d_in[1];
  const float* bq  = (const float*)d_in[2];
  const float* Wk  = (const float*)d_in[3];
  const float* bk  = (const float*)d_in[4];
  const float* Wv  = (const float*)d_in[5];
  const float* bv  = (const float*)d_in[6];
  const float* Wo  = (const float*)d_in[7];
  const float* bo  = (const float*)d_in[8];
  const float* W1  = (const float*)d_in[9];
  const float* b1  = (const float*)d_in[10];
  const float* W2  = (const float*)d_in[11];
  const float* b2  = (const float*)d_in[12];
  const float* g1  = (const float*)d_in[13];
  const float* be1 = (const float*)d_in[14];
  const float* g2  = (const float*)d_in[15];
  const float* be2 = (const float*)d_in[16];
  float* out = (float*)d_out;

  const int D = D_MODEL, F = D_FF, H = NHEAD, S = SEQ, Bb = BATCH;
  const size_t M = (size_t)Bb * S;

  uint8_t* p = (uint8_t*)d_ws;
  auto alloc = [&](size_t bytes) {
    uint8_t* r = p;
    p += (bytes + 255) & ~(size_t)255;
    return r;
  };
  bf16*  xb   = (bf16*) alloc(M * D * 2);
  bf16*  wqb  = (bf16*) alloc((size_t)D * D * 2);
  bf16*  wkb  = (bf16*) alloc((size_t)D * D * 2);
  bf16*  wvb  = (bf16*) alloc((size_t)D * D * 2);
  bf16*  wob  = (bf16*) alloc((size_t)D * D * 2);
  bf16*  w1b  = (bf16*) alloc((size_t)D * F * 2);
  bf16*  w2b  = (bf16*) alloc((size_t)F * D * 2);
  bf16*  qb   = (bf16*) alloc(M * D * 2);
  bf16*  kbuf = (bf16*) alloc(M * D * 2);
  bf16*  vbuf = (bf16*) alloc(M * D * 2);
  bf16*  ctxb = (bf16*) alloc(M * D * 2);
  float* attn = (float*)alloc(M * D * 4);
  float* hf   = (float*)alloc(M * D * 4);
  bf16*  hb   = (bf16*) alloc(M * D * 2);
  bf16*  ff1b = (bf16*) alloc(M * F * 2);
  float* ff2  = (float*)alloc(M * D * 4);

  auto cvt = [&](const float* src, bf16* dst, long n) {
    f32_to_bf16<<<1024, 256, 0, stream>>>(src, dst, n);
  };
  cvt(x,  xb,  (long)(M * D));
  cvt(Wq, wqb, (long)D * D);
  cvt(Wk, wkb, (long)D * D);
  cvt(Wv, wvb, (long)D * D);
  cvt(Wo, wob, (long)D * D);
  cvt(W1, w1b, (long)D * F);
  cvt(W2, w2b, (long)F * D);

  const dim3 blk(256);
  // Q projection with folded 1/sqrt(dk) scaling (ACT=2)
  gemm_bf16_wmma<2, false, true><<<dim3(D / 128, (int)(M / 128)), blk, 0, stream>>>(
      xb, wqb, bq, nullptr, qb,   (int)M, D, D);
  gemm_bf16_wmma<0, false, true><<<dim3(D / 128, (int)(M / 128)), blk, 0, stream>>>(
      xb, wkb, bk, nullptr, kbuf, (int)M, D, D);
  gemm_bf16_wmma<0, false, true><<<dim3(D / 128, (int)(M / 128)), blk, 0, stream>>>(
      xb, wvb, bv, nullptr, vbuf, (int)M, D, D);

  attention_wmma<<<dim3(S / 64, Bb * H), dim3(128), 0, stream>>>(
      qb, kbuf, vbuf, ctxb, S, D, H);

  gemm_bf16_wmma<0, true, false><<<dim3(D / 128, (int)(M / 128)), blk, 0, stream>>>(
      ctxb, wob, bo, attn, nullptr, (int)M, D, D);
  add_layernorm<<<dim3((int)M), blk, 0, stream>>>(x, attn, g1, be1, hf, hb, D);

  gemm_bf16_wmma<1, false, true><<<dim3(F / 128, (int)(M / 128)), blk, 0, stream>>>(
      hb, w1b, b1, nullptr, ff1b, (int)M, F, D);
  gemm_bf16_wmma<0, true, false><<<dim3(D / 128, (int)(M / 128)), blk, 0, stream>>>(
      ff1b, w2b, b2, ff2, nullptr, (int)M, D, F);
  add_layernorm<<<dim3((int)M), blk, 0, stream>>>(hf, ff2, g2, be2, out, nullptr, D);
}
